// GridNetBlock_89575837925560
// MI455X (gfx1250) — matrix operand order repaired and checked by reading, not compiled, not run on hardware
//
#include <hip/hip_runtime.h>
#include <math.h>

// ---------------------------------------------------------------------------
// GridNetBlock for MI455X (gfx1250, wave32).
// All GEMM-shaped work (LSTM input projections, LSTM recurrence, convT as
// im2col-GEMM, QKV projections, QK^T / attn@V, output projection) runs on
// v_wmma_f32_16x16x32_bf16 (bf16 inputs, f32 accumulate).
// Round 2: register-blocked 16x64 macro-tile GEMM (4x A-fragment reuse) for
// all N%64==0 shapes; LSTM recurrence stages full Whh (128 KB) in LDS.
// ---------------------------------------------------------------------------

typedef __attribute__((ext_vector_type(16))) __bf16 bf16x16;
typedef __attribute__((ext_vector_type(8)))  float  f32x8;

#define BB   4
#define CC   64
#define TT   128
#define QQ   65
#define G4   512      /* 4*H            */
#define CKK  256      /* C*EMB_KS       */
#define NL1  62       /* intra seq len  */
#define NS1  512      /* B*T            */
#define NL2  125      /* inter seq len  */
#define NS2  260      /* B*Qp           */
#define NS2P 272      /* padded to 16   */
#define EE   8
#define CDV  16
#define EF   520      /* E*F            */
#define EFP  544      /* padded K%32    */
#define CDF  1040     /* Cd*F           */
#define MATT 33280    /* B*T*Q          */
#define NTOT 2129920  /* B*C*T*Q        */

static __device__ __forceinline__ __bf16 f2bf(float f) {
  unsigned u = __builtin_bit_cast(unsigned, f);
  unsigned r = (u + 0x7FFFu + ((u >> 16) & 1u)) >> 16;
  unsigned short h = (unsigned short)r;
  return __builtin_bit_cast(__bf16, h);
}
static __device__ __forceinline__ float sigmf(float x) { return 1.0f / (1.0f + __expf(-x)); }
// A fragment (16x32 bf16): lane half, vgpr v -> K offset
static __device__ __forceinline__ int kmapA(int v, int half) { return (v >> 2) * 16 + half * 8 + (v & 3) * 2; }
// B fragment (32x16 bf16): lane half, vgpr v -> K offset
static __device__ __forceinline__ int kmapB(int v, int half) { return (v >> 2) * 16 + half * 8 + (v & 3) * 2; }

// ---------------------------------------------------------------------------
// Generic WMMA GEMM (one 16x16 tile per wave). Used for small-N shapes.
// C[M,N] = scale * (A[M,K] @ Bw[N,K]^T) + bias0 + bias1
// ---------------------------------------------------------------------------
__global__ void __launch_bounds__(256)
wmma_gemm_nt(const __bf16* __restrict__ A, const __bf16* __restrict__ Bw,
             float* __restrict__ C, int M, int N, int K,
             const float* __restrict__ bias0, const float* __restrict__ bias1,
             float scale, long long sA, long long sB, long long sC)
{
  A  += (long long)blockIdx.z * sA;
  Bw += (long long)blockIdx.z * sB;
  C  += (long long)blockIdx.z * sC;
  const int lane = threadIdx.x & 31;
  const int wave = threadIdx.x >> 5;
  const int tilesN = N >> 4;
  const long long tile = (long long)blockIdx.x * 8 + wave;
  if (tile >= (long long)(M >> 4) * tilesN) return;  // wave-uniform
  const int tm = (int)(tile / tilesN);
  const int tn = (int)(tile % tilesN);
  const int half = lane >> 4, l15 = lane & 15;
  const __bf16* arow = A  + (long long)(tm * 16 + l15) * K;
  const __bf16* brow = Bw + (long long)(tn * 16 + l15) * K;
  f32x8 acc = {0.f, 0.f, 0.f, 0.f, 0.f, 0.f, 0.f, 0.f};
  for (int k0 = 0; k0 < K; k0 += 32) {
    bf16x16 af, bfv;
#pragma unroll
    for (int v = 0; v < 8; ++v) {
      const int ka = k0 + kmapA(v, half);
      af[2 * v]     = arow[ka];
      af[2 * v + 1] = arow[ka + 1];
      const int kb = k0 + kmapB(v, half);
      bfv[2 * v]     = brow[kb];
      bfv[2 * v + 1] = brow[kb + 1];
    }
    acc = __builtin_amdgcn_wmma_f32_16x16x32_bf16(false, af, false, bfv,
                                                  (short)0, acc, false, false);
  }
  const int n_out = tn * 16 + l15;
  float badd = 0.f;
  if (bias0) badd += bias0[n_out];
  if (bias1) badd += bias1[n_out];
#pragma unroll
  for (int r = 0; r < 8; ++r) {
    const int m_out = tm * 16 + r + 8 * half;
    C[(long long)m_out * N + n_out] = acc[r] * scale + badd;
  }
}

// ---------------------------------------------------------------------------
// Register-blocked WMMA GEMM: 16x64 macro-tile per wave (4 accumulators,
// one A fragment reused 4x -> 4 back-to-back WMMAs per A load).
// Requires N % 64 == 0.
// ---------------------------------------------------------------------------
__global__ void __launch_bounds__(256)
wmma_gemm_nt_x4(const __bf16* __restrict__ A, const __bf16* __restrict__ Bw,
                float* __restrict__ C, int M, int N, int K,
                const float* __restrict__ bias0, const float* __restrict__ bias1,
                float scale, long long sA, long long sB, long long sC)
{
  A  += (long long)blockIdx.z * sA;
  Bw += (long long)blockIdx.z * sB;
  C  += (long long)blockIdx.z * sC;
  const int lane = threadIdx.x & 31;
  const int wave = threadIdx.x >> 5;
  const int nBlkN = N >> 6;  // 64-wide macro columns
  const long long mt = (long long)blockIdx.x * 8 + wave;
  if (mt >= (long long)(M >> 4) * nBlkN) return;  // wave-uniform
  const int tm = (int)(mt / nBlkN);
  const int nb = (int)(mt % nBlkN);
  const int half = lane >> 4, l15 = lane & 15;
  const __bf16* arow = A  + (long long)(tm * 16 + l15) * K;
  const __bf16* brow = Bw + (long long)(nb * 64 + l15) * K;
  f32x8 acc[4];
#pragma unroll
  for (int tt = 0; tt < 4; ++tt)
    acc[tt] = (f32x8){0.f, 0.f, 0.f, 0.f, 0.f, 0.f, 0.f, 0.f};
  for (int k0 = 0; k0 < K; k0 += 32) {
    bf16x16 af;
#pragma unroll
    for (int v = 0; v < 8; ++v) {
      const int ka = k0 + kmapA(v, half);
      af[2 * v]     = arow[ka];
      af[2 * v + 1] = arow[ka + 1];
    }
    bf16x16 bfr[4];
#pragma unroll
    for (int tt = 0; tt < 4; ++tt) {
      const __bf16* bp = brow + tt * 16 * K;
#pragma unroll
      for (int v = 0; v < 8; ++v) {
        const int kb = k0 + kmapB(v, half);
        bfr[tt][2 * v]     = bp[kb];
        bfr[tt][2 * v + 1] = bp[kb + 1];
      }
    }
#pragma unroll
    for (int tt = 0; tt < 4; ++tt)
      acc[tt] = __builtin_amdgcn_wmma_f32_16x16x32_bf16(false, af, false, bfr[tt],
                                                        (short)0, acc[tt], false, false);
  }
#pragma unroll
  for (int tt = 0; tt < 4; ++tt) {
    const int n_out = nb * 64 + tt * 16 + l15;
    float badd = 0.f;
    if (bias0) badd += bias0[n_out];
    if (bias1) badd += bias1[n_out];
#pragma unroll
    for (int r = 0; r < 8; ++r) {
      const int m_out = tm * 16 + r + 8 * half;
      C[(long long)m_out * N + n_out] = acc[tt][r] * scale + badd;
    }
  }
}

// ---------------------------------------------------------------------------
// LSTM recurrence. Gpre = X@Wih^T + bih + bhh precomputed ([2][Npad][nL][512]).
// Each block owns 16 sequences for one direction; Whh[dir] (512x128 bf16 =
// 128 KB) is staged in LDS once, h (bf16) + Z (f32) live in LDS, cell state
// in registers. Per step: Z = Gpre + h@Whh^T via WMMA
// (8 waves x 4 N-tiles x 4 K-chunks), then gate math.
// Y layout: [Npad][nL][256] (fwd | bwd concat). Static LDS = 164 KB (< 320 KB).
// ---------------------------------------------------------------------------
__global__ void __launch_bounds__(256)
lstm_recurrent(const float* __restrict__ Gpre, const __bf16* __restrict__ Whh,
               float* __restrict__ Y, int Npad, int nL)
{
  __shared__ __bf16 wsh[G4 * 128];   // 128 KB: Whh[dir], row-major [512][128]
  __shared__ __bf16 hbuf[16][128];   // 4 KB
  __shared__ float  Z[16][G4];       // 32 KB
  const int nblk = Npad >> 4;
  const int dir  = blockIdx.x / nblk;
  const int sb   = blockIdx.x % nblk;
  const int tid  = threadIdx.x;
  const int lane = tid & 31, wave = tid >> 5;
  const int half = lane >> 4, l15 = lane & 15;
  // stage Whh[dir] into LDS with b128 loads
  {
    const uint4* src = (const uint4*)(Whh + (long long)dir * G4 * 128);
    uint4* dst = (uint4*)wsh;
    for (int i = tid; i < (G4 * 128 * 2) / 16; i += 256) dst[i] = src[i];
  }
  for (int i = tid; i < 16 * 128; i += 256) ((__bf16*)hbuf)[i] = f2bf(0.f);
  float creg[8];
#pragma unroll
  for (int q = 0; q < 8; ++q) creg[q] = 0.f;
  const float* gpre = Gpre + (long long)dir * Npad * nL * G4;
  __syncthreads();
  for (int t = 0; t < nL; ++t) {
    const int step = dir ? (nL - 1 - t) : t;
#pragma unroll
    for (int nti = 0; nti < 4; ++nti) {
      const int nt = wave * 4 + nti;
      const int j  = nt * 16 + l15;
      f32x8 acc;
#pragma unroll
      for (int r = 0; r < 8; ++r) {
        const int s = r + 8 * half;
        acc[r] = gpre[((long long)(sb * 16 + s) * nL + step) * G4 + j];
      }
#pragma unroll
      for (int kc = 0; kc < 4; ++kc) {
        bf16x16 af, bfv;
#pragma unroll
        for (int v = 0; v < 8; ++v) {
          const int ka = kc * 32 + kmapA(v, half);
          af[2 * v]     = hbuf[l15][ka];
          af[2 * v + 1] = hbuf[l15][ka + 1];
          const int kb = kc * 32 + kmapB(v, half);
          bfv[2 * v]     = wsh[j * 128 + kb];
          bfv[2 * v + 1] = wsh[j * 128 + kb + 1];
        }
        acc = __builtin_amdgcn_wmma_f32_16x16x32_bf16(false, af, false, bfv,
                                                      (short)0, acc, false, false);
      }
#pragma unroll
      for (int r = 0; r < 8; ++r) Z[r + 8 * half][j] = acc[r];
    }
    __syncthreads();
    const int s = tid & 15, ub = (tid >> 4) * 8;
#pragma unroll
    for (int q = 0; q < 8; ++q) {
      const int u = ub + q;
      const float zi = Z[s][u];
      const float zf = Z[s][128 + u];
      const float zg = Z[s][256 + u];
      const float zo = Z[s][384 + u];
      const float cn = sigmf(zf) * creg[q] + sigmf(zi) * tanhf(zg);
      creg[q] = cn;
      const float hn = sigmf(zo) * tanhf(cn);
      hbuf[s][u] = f2bf(hn);
      Y[((long long)(sb * 16 + s) * nL + step) * 256 + dir * 128 + u] = hn;
    }
    __syncthreads();
  }
}

// ------------------------------ helpers ------------------------------------
__global__ void k_f32_to_bf16(const float* __restrict__ in, __bf16* __restrict__ out, long long n) {
  long long i = (long long)blockIdx.x * blockDim.x + threadIdx.x;
  const long long st = (long long)gridDim.x * blockDim.x;
  for (; i < n; i += st) out[i] = f2bf(in[i]);
}

// Wt (2H,C,4) -> Wt2[o][c2*4+kk] bf16 (64 x 1024)
__global__ void k_pack_wt(const float* __restrict__ Wt, __bf16* __restrict__ out) {
  int i = blockIdx.x * blockDim.x + threadIdx.x;
  if (i >= 64 * 1024) return;
  const int o = i >> 10, f = i & 1023, c2 = f >> 2, kk = f & 3;
  out[i] = f2bf(Wt[(c2 * 64 + o) * 4 + kk]);
}

// LayerNorm over C of (B,C,T,Q)
__global__ void k_ln_c(const float* __restrict__ X, const float* __restrict__ g,
                       const float* __restrict__ bt, float* __restrict__ Y) {
  long long i = (long long)blockIdx.x * blockDim.x + threadIdx.x;
  if (i >= (long long)BB * TT * QQ) return;
  const int pos = (int)(i % (TT * QQ));
  const int b   = (int)(i / (TT * QQ));
  const long long cs = (long long)TT * QQ;
  const float* xb = X + (long long)b * CC * cs + pos;
  float mu = 0.f;
  for (int c = 0; c < CC; ++c) mu += xb[c * cs];
  mu *= (1.f / CC);
  float var = 0.f;
  for (int c = 0; c < CC; ++c) { const float d = xb[c * cs] - mu; var += d * d; }
  var *= (1.f / CC);
  const float inv = rsqrtf(var + 1e-5f);
  float* yb = Y + (long long)b * CC * cs + pos;
  for (int c = 0; c < CC; ++c) yb[c * cs] = (xb[c * cs] - mu) * inv * g[c] + bt[c];
}

// intra unfold pack: X[n][j][c*4+kk] = LN[b,c,t,j+kk], n = b*T + t
__global__ void k_pack_intra(const float* __restrict__ L, __bf16* __restrict__ X) {
  long long i = (long long)blockIdx.x * blockDim.x + threadIdx.x;
  const long long st = (long long)gridDim.x * blockDim.x;
  const long long total = (long long)NS1 * NL1 * CKK;
  for (; i < total; i += st) {
    const int feat = (int)(i % CKK);
    const long long nj = i / CKK;
    const int j = (int)(nj % NL1);
    const int n = (int)(nj / NL1);
    const int c = feat >> 2, kk = feat & 3;
    const int b = n >> 7, t = n & 127;
    X[i] = f2bf(L[(((long long)b * CC + c) * TT + t) * QQ + j + kk]);
  }
}

// inter unfold pack: X[n][j][c*4+kk] = LN[b,c,j+kk,q], n = b*Qp + q, pad rows zero
__global__ void k_pack_inter(const float* __restrict__ L, __bf16* __restrict__ X) {
  long long i = (long long)blockIdx.x * blockDim.x + threadIdx.x;
  const long long st = (long long)gridDim.x * blockDim.x;
  const long long total = (long long)NS2P * NL2 * CKK;
  for (; i < total; i += st) {
    const int feat = (int)(i % CKK);
    const long long nj = i / CKK;
    const int j = (int)(nj % NL2);
    const int n = (int)(nj / NL2);
    float v = 0.f;
    if (n < NS2) {
      const int b = n / QQ, q = n % QQ;
      const int c = feat >> 2, kk = feat & 3;
      v = L[(((long long)b * CC + c) * TT + (j + kk)) * QQ + q];
    }
    X[i] = f2bf(v);
  }
}

// convT im2col: A[n*outL+p][c2*4+kk] = Y[n][p-kk][c2] (0 if OOB)
__global__ void k_pack_conv(const float* __restrict__ Y, __bf16* __restrict__ A,
                            int Npad, int nL, int outL) {
  long long i = (long long)blockIdx.x * blockDim.x + threadIdx.x;
  const long long st = (long long)gridDim.x * blockDim.x;
  const long long total = (long long)Npad * outL * 1024;
  for (; i < total; i += st) {
    const int feat = (int)(i % 1024);
    const long long np = i / 1024;
    const int p = (int)(np % outL);
    const int n = (int)(np / outL);
    const int c2 = feat >> 2, kk = feat & 3;
    const int j = p - kk;
    float v = 0.f;
    if (j >= 0 && j < nL) v = Y[((long long)n * nL + j) * 256 + c2];
    A[i] = f2bf(v);
  }
}

__global__ void k_res_intra(const float* __restrict__ X, const float* __restrict__ Cv,
                            float* __restrict__ R) {
  long long i = (long long)blockIdx.x * blockDim.x + threadIdx.x;
  if (i >= (long long)NTOT) return;
  const int q = (int)(i % QQ);
  long long r1 = i / QQ;
  const int t = (int)(r1 % TT);
  long long r2 = r1 / TT;
  const int c = (int)(r2 % CC);
  const int b = (int)(r2 / CC);
  R[i] = X[i] + Cv[((long long)(b * TT + t) * QQ + q) * CC + c];
}

__global__ void k_res_inter(const float* __restrict__ X, const float* __restrict__ Cv,
                            float* __restrict__ R) {
  long long i = (long long)blockIdx.x * blockDim.x + threadIdx.x;
  if (i >= (long long)NTOT) return;
  const int q = (int)(i % QQ);
  long long r1 = i / QQ;
  const int t = (int)(r1 % TT);
  long long r2 = r1 / TT;
  const int c = (int)(r2 % CC);
  const int b = (int)(r2 / CC);
  R[i] = X[i] + Cv[((long long)(b * QQ + q) * TT + t) * CC + c];
}

// inter (B,C,T,Q) -> A[(b,t,f)][c] bf16
__global__ void k_pack_att(const float* __restrict__ R, __bf16* __restrict__ A) {
  long long i = (long long)blockIdx.x * blockDim.x + threadIdx.x;
  if (i >= (long long)MATT * CC) return;
  const int c = (int)(i % CC);
  long long r = i / CC;
  const int f = (int)(r % QQ);
  long long r2 = r / QQ;
  const int t = (int)(r2 % TT);
  const int b = (int)(r2 / TT);
  A[i] = f2bf(R[(((long long)b * CC + c) * TT + t) * QQ + f]);
}

// per-(h,b,t) PReLU + LN over (E,F); output [h*B][T][EFP] bf16 (padded zeros)
__global__ void __launch_bounds__(256)
k_ln_head_qk(const float* __restrict__ R, const float* __restrict__ alpha,
             const float* __restrict__ g, const float* __restrict__ bl,
             __bf16* __restrict__ Out) {
  __shared__ float s1[256], s2[256];
  const int blk = blockIdx.x;
  const int t = blk % TT, b = (blk / TT) % BB, hh = blk / (TT * BB);
  const float a = alpha[hh];
  float acc = 0.f, acc2 = 0.f;
  for (int i = threadIdx.x; i < EF; i += 256) {
    const int e = i / QQ, f = i % QQ;
    float v = R[((long long)(b * TT + t) * QQ + f) * 32 + hh * EE + e];
    v = v >= 0.f ? v : a * v;
    acc += v; acc2 += v * v;
  }
  s1[threadIdx.x] = acc; s2[threadIdx.x] = acc2;
  __syncthreads();
  for (int o = 128; o > 0; o >>= 1) {
    if (threadIdx.x < (unsigned)o) { s1[threadIdx.x] += s1[threadIdx.x + o]; s2[threadIdx.x] += s2[threadIdx.x + o]; }
    __syncthreads();
  }
  const float mu = s1[0] * (1.f / EF);
  const float var = s2[0] * (1.f / EF) - mu * mu;
  const float inv = rsqrtf(var + 1e-5f);
  __bf16* ob = Out + ((long long)(hh * BB + b) * TT + t) * EFP;
  for (int i = threadIdx.x; i < EFP; i += 256) {
    float o = 0.f;
    if (i < EF) {
      const int e = i / QQ, f = i % QQ;
      float v = R[((long long)(b * TT + t) * QQ + f) * 32 + hh * EE + e];
      v = v >= 0.f ? v : a * v;
      o = (v - mu) * inv * g[(hh * EE + e) * QQ + f] + bl[(hh * EE + e) * QQ + f];
    }
    ob[i] = f2bf(o);
  }
}

// per-(h,b,t) PReLU + LN over (Cd,F); writes transposed Vt[h*B][Cd*F][T] bf16
__global__ void __launch_bounds__(256)
k_ln_head_v(const float* __restrict__ R, const float* __restrict__ alpha,
            const float* __restrict__ g, const float* __restrict__ bl,
            __bf16* __restrict__ Out) {
  __shared__ float s1[256], s2[256];
  const int blk = blockIdx.x;
  const int t = blk % TT, b = (blk / TT) % BB, hh = blk / (TT * BB);
  const float a = alpha[hh];
  float acc = 0.f, acc2 = 0.f;
  for (int i = threadIdx.x; i < CDF; i += 256) {
    const int cd = i / QQ, f = i % QQ;
    float v = R[((long long)(b * TT + t) * QQ + f) * 64 + hh * CDV + cd];
    v = v >= 0.f ? v : a * v;
    acc += v; acc2 += v * v;
  }
  s1[threadIdx.x] = acc; s2[threadIdx.x] = acc2;
  __syncthreads();
  for (int o = 128; o > 0; o >>= 1) {
    if (threadIdx.x < (unsigned)o) { s1[threadIdx.x] += s1[threadIdx.x + o]; s2[threadIdx.x] += s2[threadIdx.x + o]; }
    __syncthreads();
  }
  const float mu = s1[0] * (1.f / CDF);
  const float var = s2[0] * (1.f / CDF) - mu * mu;
  const float inv = rsqrtf(var + 1e-5f);
  for (int i = threadIdx.x; i < CDF; i += 256) {
    const int cd = i / QQ, f = i % QQ;
    float v = R[((long long)(b * TT + t) * QQ + f) * 64 + hh * CDV + cd];
    v = v >= 0.f ? v : a * v;
    const float o = (v - mu) * inv * g[(hh * CDV + cd) * QQ + f] + bl[(hh * CDV + cd) * QQ + f];
    Out[((long long)(hh * BB + b) * CDF + i) * TT + t] = f2bf(o);
  }
}

__global__ void __launch_bounds__(128)
k_softmax(const float* __restrict__ S, __bf16* __restrict__ A) {
  __shared__ float sh[128];
  const int row = blockIdx.x;           // nb*128 + t
  const int tid = threadIdx.x;
  const float v = S[(long long)row * 128 + tid];
  sh[tid] = v;
  __syncthreads();
  for (int o = 64; o > 0; o >>= 1) { if (tid < o) sh[tid] = fmaxf(sh[tid], sh[tid + o]); __syncthreads(); }
  const float mx = sh[0];
  __syncthreads();
  const float e = __expf(v - mx);
  sh[tid] = e;
  __syncthreads();
  for (int o = 64; o > 0; o >>= 1) { if (tid < o) sh[tid] += sh[tid + o]; __syncthreads(); }
  A[(long long)row * 128 + tid] = f2bf(e / sh[0]);
}

// Vo[h*B][T][Cd*F] -> A[(b,t,f)][h*16+cd] bf16
__global__ void k_pack_vo(const float* __restrict__ Vo, __bf16* __restrict__ A) {
  long long i = (long long)blockIdx.x * blockDim.x + threadIdx.x;
  if (i >= (long long)MATT * CC) return;
  const int k = (int)(i % CC);
  long long r = i / CC;
  const int f = (int)(r % QQ);
  long long r2 = r / QQ;
  const int t = (int)(r2 % TT);
  const int b = (int)(r2 / TT);
  const int hh = k >> 4, cd = k & 15;
  A[i] = f2bf(Vo[((long long)(hh * BB + b) * TT + t) * CDF + cd * QQ + f]);
}

// final PReLU + LN over (C,F) per (b,t), add inter residual, write (B,C,T,Q)
__global__ void __launch_bounds__(256)
k_final(const float* __restrict__ P, const float* __restrict__ ap,
        const float* __restrict__ gp, const float* __restrict__ bpl,
        const float* __restrict__ inter, float* __restrict__ out) {
  __shared__ float s1[256], s2[256];
  const int blk = blockIdx.x;
  const int t = blk % TT, b = blk / TT;
  const float a = ap[0];
  const int NE = CC * QQ;  // 4160
  float acc = 0.f, acc2 = 0.f;
  for (int i = threadIdx.x; i < NE; i += 256) {
    const int c = i / QQ, f = i % QQ;
    float v = P[((long long)(b * TT + t) * QQ + f) * CC + c];
    v = v >= 0.f ? v : a * v;
    acc += v; acc2 += v * v;
  }
  s1[threadIdx.x] = acc; s2[threadIdx.x] = acc2;
  __syncthreads();
  for (int o = 128; o > 0; o >>= 1) {
    if (threadIdx.x < (unsigned)o) { s1[threadIdx.x] += s1[threadIdx.x + o]; s2[threadIdx.x] += s2[threadIdx.x + o]; }
    __syncthreads();
  }
  const float mu = s1[0] / (float)NE;
  const float var = s2[0] / (float)NE - mu * mu;
  const float inv = rsqrtf(var + 1e-5f);
  for (int i = threadIdx.x; i < NE; i += 256) {
    const int c = i / QQ, f = i % QQ;
    float v = P[((long long)(b * TT + t) * QQ + f) * CC + c];
    v = v >= 0.f ? v : a * v;
    const float o = (v - mu) * inv * gp[c * QQ + f] + bpl[c * QQ + f];
    const long long oi = (((long long)b * CC + c) * TT + t) * QQ + f;
    out[oi] = o + inter[oi];
  }
}

// ---------------------------------------------------------------------------
extern "C" void kernel_launch(void* const* d_in, const int* in_sizes, int n_in,
                              void* d_out, int out_size, void* d_ws, size_t ws_size,
                              hipStream_t stream) {
  (void)in_sizes; (void)n_in; (void)out_size; (void)ws_size;
  const float* x      = (const float*)d_in[0];
  const float* i_lng  = (const float*)d_in[1];
  const float* i_lnb  = (const float*)d_in[2];
  const float* i_Wih  = (const float*)d_in[3];
  const float* i_Whh  = (const float*)d_in[4];
  const float* i_bih  = (const float*)d_in[5];
  const float* i_bhh  = (const float*)d_in[6];
  const float* i_Wt   = (const float*)d_in[7];
  const float* i_bt   = (const float*)d_in[8];
  const float* e_lng  = (const float*)d_in[9];
  const float* e_lnb  = (const float*)d_in[10];
  const float* e_Wih  = (const float*)d_in[11];
  const float* e_Whh  = (const float*)d_in[12];
  const float* e_bih  = (const float*)d_in[13];
  const float* e_bhh  = (const float*)d_in[14];
  const float* e_Wt   = (const float*)d_in[15];
  const float* e_bt   = (const float*)d_in[16];
  const float* Wq     = (const float*)d_in[17];
  const float* bq     = (const float*)d_in[18];
  const float* aq     = (const float*)d_in[19];
  const float* gq     = (const float*)d_in[20];
  const float* bq_ln  = (const float*)d_in[21];
  const float* Wk     = (const float*)d_in[22];
  const float* bk     = (const float*)d_in[23];
  const float* ak     = (const float*)d_in[24];
  const float* gk     = (const float*)d_in[25];
  const float* bk_ln  = (const float*)d_in[26];
  const float* Wv     = (const float*)d_in[27];
  const float* bv     = (const float*)d_in[28];
  const float* av     = (const float*)d_in[29];
  const float* gv     = (const float*)d_in[30];
  const float* bv_ln  = (const float*)d_in[31];
  const float* Wp     = (const float*)d_in[32];
  const float* bp     = (const float*)d_in[33];
  const float* apar   = (const float*)d_in[34];
  const float* gp     = (const float*)d_in[35];
  const float* bp_ln  = (const float*)d_in[36];
  float* out = (float*)d_out;

  char* ws = (char*)d_ws;
  size_t off = 0;
  auto carve = [&](size_t bytes) -> char* {
    off = (off + 255) & ~(size_t)255;
    char* p = ws + off; off += bytes; return p;
  };
  float*  lnbuf  = (float*)carve((size_t)NTOT * 4);
  float*  resb   = (float*)carve((size_t)NTOT * 4);
  float*  interb = (float*)carve((size_t)NTOT * 4);
  __bf16* Xbf    = (__bf16*)carve((size_t)34000 * 256 * 2);
  float*  Ybuf   = (float*)carve((size_t)34000 * 256 * 4);
  float*  Cconv  = (float*)carve((size_t)34816 * 64 * 4);
  __bf16* wih1   = (__bf16*)carve((size_t)2 * 512 * 256 * 2);
  __bf16* whh1   = (__bf16*)carve((size_t)2 * 512 * 128 * 2);
  __bf16* wih2   = (__bf16*)carve((size_t)2 * 512 * 256 * 2);
  __bf16* whh2   = (__bf16*)carve((size_t)2 * 512 * 128 * 2);
  __bf16* wt1    = (__bf16*)carve((size_t)64 * 1024 * 2);
  __bf16* wt2    = (__bf16*)carve((size_t)64 * 1024 * 2);
  __bf16* wqb    = (__bf16*)carve((size_t)32 * 64 * 2);
  __bf16* wkb    = (__bf16*)carve((size_t)32 * 64 * 2);
  __bf16* wvb    = (__bf16*)carve((size_t)64 * 64 * 2);
  __bf16* wpb    = (__bf16*)carve((size_t)64 * 64 * 2);
  char*   Rbase  = carve((size_t)139264000);   // reused region
  float*  Gpre   = (float*)Rbase;              // phase 1/3: LSTM pre-gemm
  __bf16* Aconv  = (__bf16*)Rbase;             // phase 2/4: im2col for convT
  size_t ro = 0;                               // phase 5: attention scratch
  auto rcarve = [&](size_t bytes) -> char* {
    ro = (ro + 255) & ~(size_t)255;
    char* p = Rbase + ro; ro += bytes; return p;
  };
  __bf16* Aattn = (__bf16*)rcarve((size_t)MATT * 64 * 2);
  float*  Qraw  = (float*)rcarve((size_t)MATT * 32 * 4);
  float*  Kraw  = (float*)rcarve((size_t)MATT * 32 * 4);
  float*  Vraw  = (float*)rcarve((size_t)MATT * 64 * 4);
  __bf16* Qf    = (__bf16*)rcarve((size_t)16 * 128 * EFP * 2);
  __bf16* Kf    = (__bf16*)rcarve((size_t)16 * 128 * EFP * 2);
  __bf16* Vt    = (__bf16*)rcarve((size_t)16 * CDF * 128 * 2);
  float*  Smat  = (float*)rcarve((size_t)16 * 128 * 128 * 4);
  __bf16* Attb  = (__bf16*)rcarve((size_t)16 * 128 * 128 * 2);
  float*  Vo    = (float*)rcarve((size_t)16 * 128 * CDF * 4);
  __bf16* Avo   = (__bf16*)rcarve((size_t)MATT * 64 * 2);
  float*  Praw  = (float*)rcarve((size_t)MATT * 64 * 4);

  auto ew = [](long long n) { return dim3((unsigned)((n + 255) / 256)); };
  // generic (N%16): small shapes
  auto gemm = [&](const __bf16* A, const __bf16* Bw, float* C, int M, int N, int K,
                  const float* b0, const float* b1, float scale,
                  long long sA, long long sB, long long sC, int batches) {
    unsigned gx = (unsigned)(((long long)(M / 16) * (N / 16) + 7) / 8);
    dim3 grid(gx, 1, (unsigned)batches);
    wmma_gemm_nt<<<grid, 256, 0, stream>>>(A, Bw, C, M, N, K, b0, b1, scale, sA, sB, sC);
  };
  // register-blocked (N%64): FLOP-heavy shapes
  auto gemm4 = [&](const __bf16* A, const __bf16* Bw, float* C, int M, int N, int K,
                   const float* b0, const float* b1, float scale,
                   long long sA, long long sB, long long sC, int batches) {
    unsigned gx = (unsigned)(((long long)(M / 16) * (N / 64) + 7) / 8);
    dim3 grid(gx, 1, (unsigned)batches);
    wmma_gemm_nt_x4<<<grid, 256, 0, stream>>>(A, Bw, C, M, N, K, b0, b1, scale, sA, sB, sC);
  };

  // --- weights -> bf16 ---
  k_f32_to_bf16<<<ew(262144), 256, 0, stream>>>(i_Wih, wih1, 262144);
  k_f32_to_bf16<<<ew(131072), 256, 0, stream>>>(i_Whh, whh1, 131072);
  k_f32_to_bf16<<<ew(262144), 256, 0, stream>>>(e_Wih, wih2, 262144);
  k_f32_to_bf16<<<ew(131072), 256, 0, stream>>>(e_Whh, whh2, 131072);
  k_f32_to_bf16<<<ew(2048),   256, 0, stream>>>(Wq, wqb, 2048);
  k_f32_to_bf16<<<ew(2048),   256, 0, stream>>>(Wk, wkb, 2048);
  k_f32_to_bf16<<<ew(4096),   256, 0, stream>>>(Wv, wvb, 4096);
  k_f32_to_bf16<<<ew(4096),   256, 0, stream>>>(Wp, wpb, 4096);
  k_pack_wt<<<ew(65536), 256, 0, stream>>>(i_Wt, wt1);
  k_pack_wt<<<ew(65536), 256, 0, stream>>>(e_Wt, wt2);

  // --- intra path ---
  k_ln_c<<<ew((long long)BB * TT * QQ), 256, 0, stream>>>(x, i_lng, i_lnb, lnbuf);
  k_pack_intra<<<ew((long long)NS1 * NL1 * CKK), 256, 0, stream>>>(lnbuf, Xbf);
  {
    const int M = NS1 * NL1;
    gemm4(Xbf, wih1, Gpre, M, G4, CKK, i_bih, i_bhh, 1.f, 0, 0, 0, 1);
    gemm4(Xbf, wih1 + 512 * 256, Gpre + (long long)M * G4, M, G4, CKK,
          i_bih + 512, i_bhh + 512, 1.f, 0, 0, 0, 1);
  }
  lstm_recurrent<<<dim3((NS1 / 16) * 2), 256, 0, stream>>>(Gpre, whh1, Ybuf, NS1, NL1);
  k_pack_conv<<<ew((long long)NS1 * QQ * 1024), 256, 0, stream>>>(Ybuf, Aconv, NS1, NL1, QQ);
  gemm4(Aconv, wt1, Cconv, NS1 * QQ, 64, 1024, i_bt, nullptr, 1.f, 0, 0, 0, 1);
  k_res_intra<<<ew(NTOT), 256, 0, stream>>>(x, Cconv, resb);

  // --- inter path ---
  k_ln_c<<<ew((long long)BB * TT * QQ), 256, 0, stream>>>(resb, e_lng, e_lnb, lnbuf);
  k_pack_inter<<<ew((long long)NS2P * NL2 * CKK), 256, 0, stream>>>(lnbuf, Xbf);
  {
    const int M = NS2P * NL2;
    gemm4(Xbf, wih2, Gpre, M, G4, CKK, e_bih, e_bhh, 1.f, 0, 0, 0, 1);
    gemm4(Xbf, wih2 + 512 * 256, Gpre + (long long)M * G4, M, G4, CKK,
          e_bih + 512, e_bhh + 512, 1.f, 0, 0, 0, 1);
  }
  lstm_recurrent<<<dim3((NS2P / 16) * 2), 256, 0, stream>>>(Gpre, whh2, Ybuf, NS2P, NL2);
  k_pack_conv<<<ew((long long)NS2P * TT * 1024), 256, 0, stream>>>(Ybuf, Aconv, NS2P, NL2, TT);
  gemm4(Aconv, wt2, Cconv, NS2P * TT, 64, 1024, e_bt, nullptr, 1.f, 0, 0, 0, 1);
  k_res_inter<<<ew(NTOT), 256, 0, stream>>>(resb, Cconv, interb);

  // --- attention ---
  k_pack_att<<<ew((long long)MATT * CC), 256, 0, stream>>>(interb, Aattn);
  gemm(Aattn, wqb, Qraw, MATT, 32, 64, bq, nullptr, 1.f, 0, 0, 0, 1);
  gemm(Aattn, wkb, Kraw, MATT, 32, 64, bk, nullptr, 1.f, 0, 0, 0, 1);
  gemm4(Aattn, wvb, Vraw, MATT, 64, 64, bv, nullptr, 1.f, 0, 0, 0, 1);
  k_ln_head_qk<<<dim3(4 * BB * TT), 256, 0, stream>>>(Qraw, aq, gq, bq_ln, Qf);
  k_ln_head_qk<<<dim3(4 * BB * TT), 256, 0, stream>>>(Kraw, ak, gk, bk_ln, Kf);
  k_ln_head_v<<<dim3(4 * BB * TT), 256, 0, stream>>>(Vraw, av, gv, bv_ln, Vt);
  gemm4(Qf, Kf, Smat, 128, 128, EFP, nullptr, nullptr, 1.f / sqrtf((float)EF),
        128LL * EFP, 128LL * EFP, 128LL * 128, 16);
  k_softmax<<<dim3(16 * 128), 128, 0, stream>>>(Smat, Attb);
  gemm(Attb, Vt, Vo, 128, CDF, 128, nullptr, nullptr, 1.f,
       128LL * 128, (long long)CDF * 128, 128LL * CDF, 16);
  k_pack_vo<<<ew((long long)MATT * CC), 256, 0, stream>>>(Vo, Avo);
  gemm4(Avo, wpb, Praw, MATT, 64, 64, bp, nullptr, 1.f, 0, 0, 0, 1);
  k_final<<<dim3(BB * TT), 256, 0, stream>>>(Praw, apar, gp, bp_ln, interb, out);
}
